// MultiTask_Agg_72859825209800
// MI455X (gfx1250) — compile-verified, matrix operand order checked by reading, not run on hardware
//
#include <hip/hip_runtime.h>
#include <hip/hip_bf16.h>

// ---------------- fixed problem shapes (from reference setup_inputs) --------
constexpr int Bc  = 8;
constexpr int Tt  = 3;
constexpr int Mm  = 4096;
constexpr int Cc  = 768;
constexpr int Hh  = 12;
constexpr int Dd  = 64;
constexpr int KK  = 256;          // capacity_factor -> top-k
constexpr int NIN = Tt + Mm;      // 4099
constexpr int N2C = 2 * Cc;       // 1536
constexpr float SCALE = 0.125f;   // D^-0.5

typedef __bf16 bf16;
typedef bf16  v2bf  __attribute__((ext_vector_type(2)));
typedef bf16  v16bf __attribute__((ext_vector_type(16)));
typedef float v8f   __attribute__((ext_vector_type(8)));
typedef int   v4i   __attribute__((ext_vector_type(4)));

// ---------------- WMMA tiling params ---------------------------------------
constexpr int BM  = 128;  // block tile rows
constexpr int BN  = 128;  // block tile cols
constexpr int BK  = 32;   // k-chunk (== WMMA bf16 K)
constexpr int SL  = 40;   // LDS row stride in bf16 elems (80 B, 16B aligned)

// ---------------- CDNA5 async global->LDS copy (ASYNCcnt-tracked) ----------
#if defined(__has_builtin)
#if __has_builtin(__builtin_amdgcn_global_load_async_to_lds_b128) && \
    __has_builtin(__builtin_amdgcn_s_wait_asynccnt)
#define USE_ASYNC_LDS 1
#endif
#endif

#ifdef USE_ASYNC_LDS
typedef v4i __attribute__((address_space(1)))* g4ptr;   // global int4*
typedef v4i __attribute__((address_space(3)))* l4ptr;   // LDS int4*
#endif

__device__ inline void asyncCopy16(void* ldsDst, const void* gSrc) {
#ifdef USE_ASYNC_LDS
  __builtin_amdgcn_global_load_async_to_lds_b128(
      (g4ptr)(void*)gSrc, (l4ptr)ldsDst, 0, 0);
#else
  *(int4*)ldsDst = *(const int4*)gSrc;
#endif
}

__device__ inline void asyncWait() {
#ifdef USE_ASYNC_LDS
  __builtin_amdgcn_s_wait_asynccnt(0);
#endif
}

// Load a 16x32 bf16 fragment (A-layout per ISA 7.12.2; B staged N-major so the
// per-VGPR K-pairs are contiguous -> ds_load_2addr_b32).
__device__ inline v16bf ldsFrag(const bf16* lds, int rc0, int stride) {
  int lane = threadIdx.x & 31;
  int sel  = lane & 15;
  int half = lane >> 4;
  const bf16* p = lds + (size_t)(rc0 + sel) * stride + 8 * half;
  v16bf f;
#pragma unroll
  for (int g = 0; g < 8; ++g) {
    int k = (g < 4) ? (2 * g) : (8 + 2 * g);   // K=0..7 then K=16..23 (+8*half)
    f[2 * g]     = p[k];
    f[2 * g + 1] = p[k + 1];
  }
  return f;
}

// ---------------- K1: fp32 -> bf16 conversions (+weight transpose), zero gates
__global__ void convert_init_kernel(const float* __restrict__ x,
                                    const float* __restrict__ kv_w,
                                    const float* __restrict__ ew,
                                    bf16* feat16, bf16* kvw16T, bf16* ew16T,
                                    float* gates) {
  size_t i      = (size_t)blockIdx.x * blockDim.x + threadIdx.x;
  size_t stride = (size_t)gridDim.x * blockDim.x;
  const size_t nFeat = (size_t)Bc * Mm * Cc;
  const size_t nKvw  = (size_t)Cc * N2C;
  const size_t nEw   = (size_t)Tt * Cc * Cc;
  const size_t nG    = (size_t)Bc * Tt * Mm * Hh;
  for (size_t e = i; e < nFeat; e += stride) {
    size_t bm = e / Cc, c = e % Cc;
    size_t b = bm / Mm, m = bm % Mm;
    feat16[e] = (bf16)x[(b * NIN + Tt + m) * Cc + c];
  }
  // kv_w (C x 2C) -> kvw16T (2C x C), K-contiguous for async B128 staging
  for (size_t e = i; e < nKvw; e += stride) {
    size_t n = e / Cc, k = e % Cc;
    kvw16T[e] = (bf16)kv_w[k * N2C + n];
  }
  // experts_w (T x C x C) -> ew16T (T x C(out) x C(in))
  for (size_t e = i; e < nEw; e += stride) {
    size_t t = e / ((size_t)Cc * Cc), r = e % ((size_t)Cc * Cc);
    size_t n = r / Cc, k = r % Cc;
    ew16T[e] = (bf16)ew[(t * Cc + k) * Cc + n];
  }
  for (size_t e = i; e < nG; e += stride) gates[e] = 0.f;
}

// ---------------- K2: q[b,t,:] = x[b,t,:] @ qs_w[t] + qs_b[t] --------------
__global__ __launch_bounds__(256) void q_kernel(const float* __restrict__ x,
                                                const float* __restrict__ qs_w,
                                                const float* __restrict__ qs_b,
                                                float* qws) {
  __shared__ float xr[Cc];
  int blk = blockIdx.x;            // b*T + t
  int t = blk % Tt, b = blk / Tt;
  int tid = threadIdx.x;
  for (int c = tid; c < Cc; c += 256) xr[c] = x[((size_t)b * NIN + t) * Cc + c];
  __syncthreads();
  for (int d = tid; d < Cc; d += 256) {
    float s = qs_b[t * Cc + d];
    const float* wp = qs_w + (size_t)t * Cc * Cc + d;
#pragma unroll 4
    for (int c = 0; c < Cc; ++c) s += xr[c] * wp[(size_t)c * Cc];
    qws[(size_t)blk * Cc + d] = s;
  }
}

// ---------------- K3: kv GEMM (bf16 WMMA), double-buffered async pipeline --
__global__ __launch_bounds__(256) void kv_gemm_kernel(
    const bf16* __restrict__ feat16, const bf16* __restrict__ kvw16T,
    const float* __restrict__ kv_b, bf16* k16, bf16* v16) {
  __shared__ __align__(16) bf16 At[2][BM * SL];
  __shared__ __align__(16) bf16 Bt[2][BN * SL];
  int tid = threadIdx.x;
  int rowTile = blockIdx.x;                    // B*M/128 = 256
  int nTile   = blockIdx.y;                    // 12
  int b     = (rowTile * BM) / Mm;
  int mBase = (rowTile * BM) % Mm;
  int nBase = nTile * BN;
  int wave = tid >> 5;
  int wm = wave & 3;                           // 4 waves x 32 rows
  int wn = wave >> 2;                          // 2 waves x 64 cols
  v8f acc[2][4];
#pragma unroll
  for (int i = 0; i < 2; ++i)
#pragma unroll
    for (int j = 0; j < 4; ++j)
#pragma unroll
      for (int r = 0; r < 8; ++r) acc[i][j][r] = 0.f;

  const bf16* Ag  = feat16 + (size_t)(b * Mm + mBase) * Cc;
  const bf16* BgT = kvw16T + (size_t)nBase * Cc;

  auto issue = [&](int kc, int buf) {
    // A: 128 rows x 32 k = 512 x 16B transfers (K-contiguous)
    for (int e = tid; e < 512; e += 256) {
      int m = e >> 2, ko = (e & 3) * 8;
      asyncCopy16(&At[buf][m * SL + ko], Ag + (size_t)m * Cc + kc + ko);
    }
    // B (pre-transposed, N-major): 128 cols x 32 k
    for (int e = tid; e < 512; e += 256) {
      int n = e >> 2, ko = (e & 3) * 8;
      asyncCopy16(&Bt[buf][n * SL + ko], BgT + (size_t)n * Cc + kc + ko);
    }
  };

  issue(0, 0);
  int cur = 0;
  for (int kc = 0; kc < Cc; kc += BK) {
    asyncWait();
    __syncthreads();                       // tiles[cur] ready, prev reads done
    if (kc + BK < Cc) issue(kc + BK, cur ^ 1);
    v16bf afr[2], bfr[4];
#pragma unroll
    for (int i = 0; i < 2; ++i) afr[i] = ldsFrag(At[cur], wm * 32 + i * 16, SL);
#pragma unroll
    for (int j = 0; j < 4; ++j) bfr[j] = ldsFrag(Bt[cur], wn * 64 + j * 16, SL);
#pragma unroll
    for (int i = 0; i < 2; ++i)
#pragma unroll
      for (int j = 0; j < 4; ++j)
        acc[i][j] = __builtin_amdgcn_wmma_f32_16x16x32_bf16(
            false, afr[i], false, bfr[j], (short)0, acc[i][j], false, false);
    cur ^= 1;
  }
  // epilogue: +bias, scatter to k16/v16 (B,H,M,D) bf16
  int lane = tid & 31;
  int nl = lane & 15, half = lane >> 4;
#pragma unroll
  for (int i = 0; i < 2; ++i)
#pragma unroll
    for (int j = 0; j < 4; ++j) {
      int ng = nBase + wn * 64 + j * 16 + nl;
      float bias = kv_b[ng];
      bool isV = ng >= Cc;
      int nc = isV ? ng - Cc : ng;
      int hh = nc >> 6, dd = nc & 63;
      bf16* arr = isV ? v16 : k16;
#pragma unroll
      for (int r = 0; r < 8; ++r) {
        int mg = mBase + wm * 32 + i * 16 + r + 8 * half;
        float val = acc[i][j][r] + bias;
        arr[(((size_t)(b * Hh + hh) * Mm) + mg) * Dd + dd] = (bf16)val;
      }
    }
}

// ---------------- K4: scores -> top-k -> softmax -> gates + attn_token -----
__global__ __launch_bounds__(256) void attn_topk_kernel(
    const float* __restrict__ qws, const bf16* __restrict__ k16,
    const bf16* __restrict__ v16, float* gates, float* attnTok) {
  __shared__ float scores[Mm];
  __shared__ float redv[256];
  __shared__ int   redi[256];
  __shared__ float topw[KK];
  __shared__ int   topidx[KK];
  __shared__ float qrow[Dd];
  const float NEGINF = -3.402823e38f;
  int blk = blockIdx.x;                     // (b*H + h)*T + t
  int t = blk % Tt;
  int h = (blk / Tt) % Hh;
  int b = blk / (Tt * Hh);
  int tid = threadIdx.x;
  if (tid < Dd) qrow[tid] = qws[(size_t)(b * Tt + t) * Cc + h * Dd + tid];
  __syncthreads();
  const bf16* kb = k16 + (size_t)(b * Hh + h) * Mm * Dd;
  for (int m = tid; m < Mm; m += 256) {
    const bf16* kr = kb + (size_t)m * Dd;
    float s = 0.f;
#pragma unroll 8
    for (int d = 0; d < Dd; ++d) s += qrow[d] * (float)kr[d];
    scores[m] = s * SCALE;
  }
  __syncthreads();
  // iterative argmax top-k (indices unique by construction)
  for (int it = 0; it < KK; ++it) {
    float bv = NEGINF; int bi = 0;
    for (int m = tid; m < Mm; m += 256) {
      float s = scores[m];
      if (s > bv || (s == bv && m < bi)) { bv = s; bi = m; }
    }
    redv[tid] = bv; redi[tid] = bi;
    __syncthreads();
    for (int off = 128; off > 0; off >>= 1) {
      if (tid < off) {
        float ov = redv[tid + off]; int oi = redi[tid + off];
        if (ov > redv[tid] || (ov == redv[tid] && oi < redi[tid])) {
          redv[tid] = ov; redi[tid] = oi;
        }
      }
      __syncthreads();
    }
    if (tid == 0) {
      topw[it] = redv[0]; topidx[it] = redi[0];
      scores[redi[0]] = NEGINF;
    }
    __syncthreads();
  }
  // softmax over topv (topw[0] is the max: selection is descending)
  float mx = topw[0];
  float w = __expf(topw[tid] - mx);
  redv[tid] = w;
  __syncthreads();
  for (int off = 128; off > 0; off >>= 1) {
    if (tid < off) redv[tid] += redv[tid + off];
    __syncthreads();
  }
  float inv = 1.f / redv[0];
  __syncthreads();
  w *= inv;
  topw[tid] = w;
  __syncthreads();
  // scatter gate for the expert path
  {
    int m = topidx[tid];
    gates[((size_t)(b * Tt + t) * Mm + m) * Hh + h] = w;
  }
  // attn_token[b,t,h*D+d] = sum_j w_j * v[b,h,idx_j,d]
  if (tid < Dd) {
    const bf16* vb = v16 + (size_t)(b * Hh + h) * Mm * Dd;
    float acc = 0.f;
#pragma unroll 4
    for (int j = 0; j < KK; ++j)
      acc += topw[j] * (float)vb[(size_t)topidx[j] * Dd + tid];
    attnTok[(size_t)(b * Tt + t) * Cc + h * Dd + tid] = acc;
  }
}

// ---------------- K5: token_output rows of out -----------------------------
__global__ __launch_bounds__(256) void token_out_kernel(
    const float* __restrict__ attnTok, const float* __restrict__ ew,
    float* out) {
  __shared__ float xr[Cc];
  int blk = blockIdx.x;            // b*T + t
  int t = blk % Tt, b = blk / Tt;
  int tid = threadIdx.x;
  for (int c = tid; c < Cc; c += 256) xr[c] = attnTok[(size_t)blk * Cc + c];
  __syncthreads();
  for (int d = tid; d < Cc; d += 256) {
    float s = 0.f;
    const float* wp = ew + (size_t)t * Cc * Cc + d;
#pragma unroll 4
    for (int c = 0; c < Cc; ++c) s += xr[c] * wp[(size_t)c * Cc];
    out[((size_t)b * NIN + t) * Cc + d] = s;
  }
}

// ---------------- K6: gated expert GEMM (bf16 WMMA), sum over t ------------
__global__ __launch_bounds__(256) void moe_gemm_kernel(
    const bf16* __restrict__ feat16, const bf16* __restrict__ ew16T,
    const float* __restrict__ gates, float* out) {
  __shared__ __align__(16) bf16 At[BM * SL];
  __shared__ __align__(16) bf16 Bt[BN * SL];
  __shared__ float gcache[BM];
  int tid = threadIdx.x;
  int rowTile = blockIdx.x;                    // B*M/128 = 256
  int nTile   = blockIdx.y;                    // 6
  int b     = (rowTile * BM) / Mm;
  int mBase = (rowTile * BM) % Mm;
  int nBase = nTile * BN;
  int wave = tid >> 5;
  int wm = wave & 3;
  int wn = wave >> 2;
  v8f acc[2][4];
#pragma unroll
  for (int i = 0; i < 2; ++i)
#pragma unroll
    for (int j = 0; j < 4; ++j)
#pragma unroll
      for (int r = 0; r < 8; ++r) acc[i][j][r] = 0.f;

  const bf16* Ag = feat16 + (size_t)(b * Mm + mBase) * Cc;
  for (int t = 0; t < Tt; ++t) {
    const float* grow = gates + ((size_t)(b * Tt + t) * Mm + mBase) * Hh;
    const bf16* BgT = ew16T + (size_t)t * Cc * Cc + (size_t)nBase * Cc;
    for (int kc = 0; kc < Cc; kc += BK) {
      int h = kc >> 6;   // 32-wide chunk never straddles a 64-wide head block
      __syncthreads();                        // prev tile reads complete
      if (tid < BM) gcache[tid] = grow[(size_t)tid * Hh + h];
      // B tile via async DMA (overlaps the gated-A staging below)
      for (int e = tid; e < 512; e += 256) {
        int n = e >> 2, ko = (e & 3) * 8;
        asyncCopy16(&Bt[n * SL + ko], BgT + (size_t)n * Cc + kc + ko);
      }
      __syncthreads();                        // gcache visible
      // A: gated feature tile, packed dword loads, converted to bf16
      for (int e = tid; e < BM * BK / 2; e += 256) {
        int m = e >> 4, kp = e & 15;
        v2bf p = *(const v2bf*)(Ag + (size_t)m * Cc + kc + 2 * kp);
        float g = gcache[m];
        p[0] = (bf16)((float)p[0] * g);
        p[1] = (bf16)((float)p[1] * g);
        *(v2bf*)&At[m * SL + 2 * kp] = p;
      }
      asyncWait();
      __syncthreads();                        // A stores + B async complete
      v16bf afr[2], bfr[4];
#pragma unroll
      for (int i = 0; i < 2; ++i) afr[i] = ldsFrag(At, wm * 32 + i * 16, SL);
#pragma unroll
      for (int j = 0; j < 4; ++j) bfr[j] = ldsFrag(Bt, wn * 64 + j * 16, SL);
#pragma unroll
      for (int i = 0; i < 2; ++i)
#pragma unroll
        for (int j = 0; j < 4; ++j)
          acc[i][j] = __builtin_amdgcn_wmma_f32_16x16x32_bf16(
              false, afr[i], false, bfr[j], (short)0, acc[i][j], false, false);
    }
  }
  // store feature_output rows (out rows T..T+M-1)
  int lane = tid & 31;
  int nl = lane & 15, half = lane >> 4;
#pragma unroll
  for (int i = 0; i < 2; ++i)
#pragma unroll
    for (int j = 0; j < 4; ++j) {
      int ng = nBase + wn * 64 + j * 16 + nl;
#pragma unroll
      for (int r = 0; r < 8; ++r) {
        int mg = mBase + wm * 32 + i * 16 + r + 8 * half;
        out[((size_t)b * NIN + Tt + mg) * Cc + ng] = acc[i][j][r];
      }
    }
}

// ---------------- host: workspace carve + launch sequence ------------------
extern "C" void kernel_launch(void* const* d_in, const int* in_sizes, int n_in,
                              void* d_out, int out_size, void* d_ws,
                              size_t ws_size, hipStream_t stream) {
  const float* x    = (const float*)d_in[0];
  const float* qs_w = (const float*)d_in[1];
  const float* qs_b = (const float*)d_in[2];
  const float* kv_w = (const float*)d_in[3];
  const float* kv_b = (const float*)d_in[4];
  const float* ew   = (const float*)d_in[5];
  float* out = (float*)d_out;

  char* ws = (char*)d_ws;
  size_t off = 0;
  auto take = [&](size_t bytes) {
    char* p = ws + off;
    off = (off + bytes + 255) & ~(size_t)255;
    return p;
  };
  bf16*  feat16  = (bf16*)take((size_t)Bc * Mm * Cc * sizeof(bf16));
  bf16*  kvw16T  = (bf16*)take((size_t)Cc * N2C * sizeof(bf16));
  bf16*  ew16T   = (bf16*)take((size_t)Tt * Cc * Cc * sizeof(bf16));
  bf16*  k16     = (bf16*)take((size_t)Bc * Hh * Mm * Dd * sizeof(bf16));
  bf16*  v16     = (bf16*)take((size_t)Bc * Hh * Mm * Dd * sizeof(bf16));
  float* qws     = (float*)take((size_t)Bc * Tt * Cc * sizeof(float));
  float* gates   = (float*)take((size_t)Bc * Tt * Mm * Hh * sizeof(float));
  float* attnTok = (float*)take((size_t)Bc * Tt * Cc * sizeof(float));
  (void)ws_size; (void)in_sizes; (void)n_in; (void)out_size;

  // 1) bf16 conversions (+ weight transposes for async K-contiguous staging)
  convert_init_kernel<<<4096, 256, 0, stream>>>(x, kv_w, ew, feat16, kvw16T,
                                                ew16T, gates);
  // 2) q projection (tiny)
  q_kernel<<<Bc * Tt, 256, 0, stream>>>(x, qs_w, qs_b, qws);
  // 3) kv GEMM: (B*M,C)x(C,2C) -> k16,v16 (double-buffered async pipeline)
  kv_gemm_kernel<<<dim3(Bc * Mm / BM, N2C / BN), 256, 0, stream>>>(
      feat16, kvw16T, kv_b, k16, v16);
  // 4) scores + top-k + softmax + gates + attn_token
  attn_topk_kernel<<<Bc * Hh * Tt, 256, 0, stream>>>(qws, k16, v16, gates,
                                                     attnTok);
  // 5) token_output rows
  token_out_kernel<<<Bc * Tt, 256, 0, stream>>>(attnTok, ew, out);
  // 6) gated expert GEMM, summed over t -> feature rows
  moe_gemm_kernel<<<dim3(Bc * Mm / BM, Cc / BN), 256, 0, stream>>>(
      feat16, ew16T, gates, out);
}
